// Decoder_30176440221863
// MI455X (gfx1250) — matrix operand order repaired and checked by reading, not compiled, not run on hardware
//
#include <hip/hip_runtime.h>

// ---------------------------------------------------------------------------
// Problem constants (match reference)
// ---------------------------------------------------------------------------
#define LNUM   4
#define HNUM   16
#define DMODEL 1024
#define DHEAD  64
#define BATCH  2
#define SEQ    2048
#define MROWS  (BATCH * SEQ)   // 4096 flattened (b,s) rows
#define VOCAB  32000

typedef __attribute__((ext_vector_type(16))) __bf16 v16bf;
typedef __attribute__((ext_vector_type(8)))  __bf16 v8bf;
typedef __attribute__((ext_vector_type(8)))  float  v8f;
typedef __attribute__((ext_vector_type(4)))  unsigned int u32x4;
typedef __attribute__((ext_vector_type(8)))  int i32x8;
typedef __attribute__((ext_vector_type(4)))  int i32x4;

// ---------------------------------------------------------------------------
// WMMA helper (CDNA5 wave32, 16x16x32 bf16 -> f32 accumulate)
// ---------------------------------------------------------------------------
__device__ __forceinline__ v8f wmma_bf16(v16bf a, v16bf b, v8f c) {
    return __builtin_amdgcn_wmma_f32_16x16x32_bf16(false, a, false, b,
                                                   (short)0, c, false, false);
}

// A fragment: 16x32 bf16, row-major source, leading dim ld.
// A layout: lanes 0-15 M=0..15 K=0..15, lanes 16-31 K=16..31,
// sequential halves (matches B).  Two 16B loads.
__device__ __forceinline__ v16bf load_a_frag(const __bf16* __restrict__ src,
                                             int ld, int lane) {
    int m  = lane & 15;
    int kb = (lane >> 4) * 16;
    union { v16bf v; v8bf h[2]; } u;
    u.h[0] = *(const v8bf*)(src + (size_t)m * ld + kb);
    u.h[1] = *(const v8bf*)(src + (size_t)m * ld + kb + 8);
    return u.v;
}

// B fragment: 32x16 bf16 where element (k,n) = srcT[n*ld + k] (N-major).
// B layout (this code, same as A): lanes 0-15 K=0..15, lanes 16-31 K=16..31, sequential halves.
__device__ __forceinline__ v16bf load_b_frag(const __bf16* __restrict__ srcT,
                                             int ld, int lane) {
    int n  = lane & 15;
    int kb = (lane >> 4) * 16;
    union { v16bf v; v8bf h[2]; } u;
    u.h[0] = *(const v8bf*)(srcT + (size_t)n * ld + kb);
    u.h[1] = *(const v8bf*)(srcT + (size_t)n * ld + kb + 8);
    return u.v;
}

// ---------------------------------------------------------------------------
// Tensor Data Mover: 2-D tile load Global -> LDS (CDNA5 ISA).
// tile: tile_n rows of tile_k bf16 elements, row stride = strideElems.
// D# group0: count=1, lds_addr, 57-bit global addr, type=2 ("image").
// D# group1: data_size=1 (2B), huge tensor dims (no OOB), tile dims, stride.
// ---------------------------------------------------------------------------
__device__ __forceinline__ void tdm_load_tile_2d(void* lds_dst,
                                                 const void* gsrc,
                                                 unsigned tile_k,
                                                 unsigned tile_n,
                                                 unsigned strideElems) {
    // Flat LDS pointer: aperture in [63:32], LDS byte offset in [31:0] (ISA §10.2)
    unsigned lds_addr = (unsigned)(unsigned long long)(uintptr_t)lds_dst;
    unsigned long long ga = (unsigned long long)(uintptr_t)gsrc;
    u32x4 g0;
    g0[0] = 1u;                                            // count=1
    g0[1] = lds_addr;                                      // lds_addr
    g0[2] = (unsigned)(ga & 0xFFFFFFFFu);                  // global_addr[31:0]
    g0[3] = (unsigned)((ga >> 32) & 0x01FFFFFFu) | (2u << 30);  // addr[56:32], type=2
    i32x8 g1;
    g1[0] = 0x00010000;                    // data_size=1 (2 bytes), wg_mask=0
    g1[1] = 0;                             // abar=0, tensor_dim0[15:0]=0
    g1[2] = 0x00000010;                    // tensor_dim0=0x100000, tdim1 lo=0
    g1[3] = (int)(tile_k << 16) | 0x10;    // tensor_dim1=0x100000, tile_dim0
    g1[4] = (int)tile_n;                   // tile_dim1 ; tile_dim2=0
    g1[5] = (int)strideElems;              // tensor_dim0_stride[31:0]
    g1[6] = 0;                             // stride hi, dim1_stride lo
    g1[7] = 0;
    i32x4 z4 = {0, 0, 0, 0};
#if defined(__clang_major__) && (__clang_major__ >= 23)
    i32x8 z8 = {0, 0, 0, 0, 0, 0, 0, 0};
    __builtin_amdgcn_tensor_load_to_lds(g0, g1, z4, z4, z8, 0);
#else
    __builtin_amdgcn_tensor_load_to_lds(g0, g1, z4, z4, 0);
#endif
}

// ---------------------------------------------------------------------------
// Elementwise converts
// ---------------------------------------------------------------------------
__global__ void x_init_kernel(const float* __restrict__ X,
                              float* __restrict__ Xcur,
                              __bf16* __restrict__ Xb, int n) {
    int i = blockIdx.x * blockDim.x + threadIdx.x;
    if (i < n) {
        float v = X[i];
        Xcur[i] = v;
        Xb[i]   = (__bf16)v;
    }
}

// in: [batch][rows][cols] fp32 -> out: [batch][cols][rows] bf16
__global__ void transpose_cvt_kernel(const float* __restrict__ in,
                                     __bf16* __restrict__ out,
                                     int rows, int cols) {
    size_t base = (size_t)blockIdx.y * rows * cols;
    int total = rows * cols;
    for (int i = blockIdx.x * blockDim.x + threadIdx.x; i < total;
         i += gridDim.x * blockDim.x) {
        int r = i / cols, c = i % cols;
        out[base + (size_t)c * rows + r] = (__bf16)in[base + i];
    }
}

// ---------------------------------------------------------------------------
// WMMA GEMM with TDM-staged weights:
//   C[M,N] = A[M,K] * B[K,N] (+bias), BT is N-major (BT[n*K+k]).
// Block = 256 threads = 8 waves; each wave computes a 16x64 strip
// (block tile 128x64).  The shared 64x32 bf16 weight tile per K-step is
// DMA'd into double-buffered LDS by the Tensor Data Mover (wave 0 issues
// tensor_load_to_lds, s_wait_tensorcnt, barrier), pipelined one tile ahead.
// grid = (M/128, N/64, heads).
// ---------------------------------------------------------------------------
enum { MODE_QK = 0, MODE_V = 1, MODE_FFN = 2 };

template <int MODE>
__global__ __launch_bounds__(256) void gemm_bf16_kernel(
    const __bf16* __restrict__ A, const __bf16* __restrict__ BT,
    const float* __restrict__ bias, void* __restrict__ Cout,
    int M, int N, int Kd,
    size_t headStrideBT, size_t headStrideC, int headStrideBias,
    float outScale) {
    __shared__ __bf16 sB[2][64 * 32];   // [buf][n][k] : 4KB per buffer

    int lane = threadIdx.x & 31;
    int wave = threadIdx.x >> 5;
    int m0 = blockIdx.x * 128 + wave * 16;
    int n0 = blockIdx.y * 64;
    int h  = blockIdx.z;
    // weight tile base: rows n0..n0+63 of BT for this head
    const __bf16* bt = BT + (size_t)h * headStrideBT + (size_t)n0 * Kd;

    // prime the pipeline: tile for k0 = 0 into buffer 0
    if (wave == 0)
        tdm_load_tile_2d(&sB[0][0], bt, 32, 64, (unsigned)Kd);

    v8f acc[4] = {v8f{}, v8f{}, v8f{}, v8f{}};
    int buf = 0;
    for (int k0 = 0; k0 < Kd; k0 += 32) {
        if (wave == 0)
            __builtin_amdgcn_s_wait_tensorcnt(0);   // tile k0 landed in LDS
        __syncthreads();
        if (wave == 0 && (k0 + 32) < Kd)            // prefetch next tile (TDM)
            tdm_load_tile_2d(&sB[buf ^ 1][0], bt + k0 + 32, 32, 64,
                             (unsigned)Kd);
        if (k0 + 64 < Kd)                           // global_prefetch for A
            __builtin_prefetch(A + (size_t)m0 * Kd + k0 + 64, 0, 1);

        // preload A fragment + all four B fragments, then issue 4 WMMAs
        v16bf a = load_a_frag(A + (size_t)m0 * Kd + k0, Kd, lane);
        v16bf b[4];
        {
            int n = lane & 15, kb = (lane >> 4) * 16;
#pragma unroll
            for (int nt = 0; nt < 4; ++nt) {
                union { v16bf v; v8bf h2[2]; } u;
                const __bf16* src = &sB[buf][(nt * 16 + n) * 32 + kb];
                u.h2[0] = *(const v8bf*)(src);
                u.h2[1] = *(const v8bf*)(src + 8);
                b[nt] = u.v;
            }
        }
#pragma unroll
        for (int nt = 0; nt < 4; ++nt)
            acc[nt] = wmma_bf16(a, b[nt], acc[nt]);
        buf ^= 1;
    }

    // C layout: VGPR v, lane l -> row = v + 8*(l>>4), col = (l&15)
    int cn  = lane & 15;
    int rhi = (lane >> 4) * 8;
#pragma unroll
    for (int nt = 0; nt < 4; ++nt) {
        int n = n0 + nt * 16 + cn;
        float bb = bias[h * headStrideBias + n];
#pragma unroll
        for (int v = 0; v < 8; ++v) {
            int row = m0 + rhi + v;
            float val = acc[nt][v] + bb;
            if constexpr (MODE == MODE_QK) {
                __bf16* outp = (__bf16*)Cout + (size_t)h * headStrideC;
                outp[(size_t)row * N + n] = (__bf16)(val * outScale);
            } else if constexpr (MODE == MODE_V) {
                __bf16* outp = (__bf16*)Cout + (size_t)h * headStrideC;
                outp[(size_t)n * M + row] = (__bf16)val;   // transposed store
            } else {  // MODE_FFN: fp32 + relu
                float* outp = (float*)Cout;
                outp[(size_t)row * N + n] = val > 0.f ? val : 0.f;
            }
        }
    }
}

// ---------------------------------------------------------------------------
// Attention: one wave per (h, b, 16-query tile).  Flash-style online softmax
// over ALL 2048 keys; tril mask applied only to the V accumulation (faithful
// to the reference's softmax-then-mask bug).  Q pre-scaled by 1/sqrt(DH).
// ---------------------------------------------------------------------------
__global__ __launch_bounds__(32) void attn_kernel(
    const __bf16* __restrict__ Qb, const __bf16* __restrict__ Kb,
    const __bf16* __restrict__ Vt, float* __restrict__ Z) {
    __shared__ float  sS[16][32];
    __shared__ __bf16 sP[16][32];
    __shared__ float  sM[16], sL[16], sC[16];

    int lane = threadIdx.x;
    int bid  = blockIdx.x;
    int qt = bid & 127;          // SEQ/16 query tiles
    int b  = (bid >> 7) & 1;
    int h  = bid >> 8;
    int mb = b * SEQ;
    int q0 = qt * 16;

    const __bf16* Qh = Qb + ((size_t)h * MROWS + mb + q0) * DHEAD;
    const __bf16* Kh = Kb + ((size_t)h * MROWS + mb) * DHEAD;
    const __bf16* Vh = Vt + (size_t)h * DHEAD * MROWS + mb;  // [dh][MROWS]

    v16bf qa0 = load_a_frag(Qh, DHEAD, lane);       // dh 0..31
    v16bf qa1 = load_a_frag(Qh + 32, DHEAD, lane);  // dh 32..63
    v8f O[4] = {v8f{}, v8f{}, v8f{}, v8f{}};

    if (lane < 16) { sM[lane] = -1e30f; sL[lane] = 0.f; }
    __syncthreads();

    int cn  = lane & 15;
    int rhi = (lane >> 4) * 8;

    for (int it = 0; it < SEQ / 32; ++it) {
        int k32 = it * 32;
        // --- preload all K-fragments, then 4 WMMAs (lets loads clause up) ---
        v16bf bk00 = load_b_frag(Kh + (size_t)k32 * DHEAD, DHEAD, lane);
        v16bf bk01 = load_b_frag(Kh + (size_t)k32 * DHEAD + 32, DHEAD, lane);
        v16bf bk10 = load_b_frag(Kh + (size_t)(k32 + 16) * DHEAD, DHEAD, lane);
        v16bf bk11 = load_b_frag(Kh + (size_t)(k32 + 16) * DHEAD + 32, DHEAD, lane);
        v8f s0 = v8f{}, s1 = v8f{};
        s0 = wmma_bf16(qa0, bk00, s0);
        s0 = wmma_bf16(qa1, bk01, s0);
        s1 = wmma_bf16(qa0, bk10, s1);
        s1 = wmma_bf16(qa1, bk11, s1);

#pragma unroll
        for (int v = 0; v < 8; ++v) {
            sS[rhi + v][cn]      = s0[v];
            sS[rhi + v][16 + cn] = s1[v];
        }
        // V fragments don't depend on P: issue loads now, overlap softmax
        v16bf bv[4];
#pragma unroll
        for (int nt = 0; nt < 4; ++nt)
            bv[nt] = load_b_frag(Vh + (size_t)(nt * 16) * MROWS + k32,
                                 MROWS, lane);
        __syncthreads();

        // --- online softmax bookkeeping, one lane per query row ---
        if (lane < 16) {
            int r = lane;
            float mNew = sM[r];
            for (int j = 0; j < 32; ++j) mNew = fmaxf(mNew, sS[r][j]);
            float corr = __expf(sM[r] - mNew);
            float lsum = sL[r] * corr;
            int qAbs = q0 + r;
            for (int j = 0; j < 32; ++j) {
                float p = __expf(sS[r][j] - mNew);
                lsum += p;  // normalize over ALL keys (faithful bug)
                sP[r][j] = (__bf16)((k32 + j <= qAbs) ? p : 0.f);  // tril mask
            }
            sM[r] = mNew; sL[r] = lsum; sC[r] = corr;
        }
        __syncthreads();

        // --- rescale O by per-row correction ---
#pragma unroll
        for (int v = 0; v < 8; ++v) {
            float c = sC[rhi + v];
#pragma unroll
            for (int nt = 0; nt < 4; ++nt) O[nt][v] *= c;
        }

        // --- O += P(16x32) x V(32x64), P A-fragment from LDS ---
        v16bf pa;
        {
            int m = lane & 15, kb = (lane >> 4) * 16;
            union { v16bf v; v8bf h2[2]; } u;
            u.h2[0] = *(const v8bf*)(&sP[m][kb]);
            u.h2[1] = *(const v8bf*)(&sP[m][kb + 8]);
            pa = u.v;
        }
#pragma unroll
        for (int nt = 0; nt < 4; ++nt)
            O[nt] = wmma_bf16(pa, bv[nt], O[nt]);
        __syncthreads();  // protect sS/sP reuse next iteration
    }

    if (lane < 16) sC[lane] = 1.f / sL[lane];
    __syncthreads();
#pragma unroll
    for (int v = 0; v < 8; ++v) {
        float inv = sC[rhi + v];
        int row = mb + q0 + rhi + v;
#pragma unroll
        for (int nt = 0; nt < 4; ++nt)
            Z[(size_t)row * DMODEL + h * DHEAD + nt * 16 + cn] = O[nt][v] * inv;
    }
}

// ---------------------------------------------------------------------------
// X = (X + R - mu) / var  (reference divides by variance, no epsilon),
// also emits bf16 copy for the next GEMM.  One block per row of 1024.
// ---------------------------------------------------------------------------
__global__ __launch_bounds__(256) void add_norm_kernel(
    const float* __restrict__ Xin, const float* __restrict__ R,
    float* __restrict__ Xout, __bf16* __restrict__ Xb) {
    __shared__ float red[256];
    int row = blockIdx.x, t = threadIdx.x;
    const float* x = Xin + (size_t)row * DMODEL;
    const float* r = R + (size_t)row * DMODEL;
    float y[4]; float s = 0.f;
#pragma unroll
    for (int j = 0; j < 4; ++j) { y[j] = x[t + 256 * j] + r[t + 256 * j]; s += y[j]; }
    red[t] = s; __syncthreads();
    for (int o = 128; o > 0; o >>= 1) { if (t < o) red[t] += red[t + o]; __syncthreads(); }
    float mu = red[0] / (float)DMODEL; __syncthreads();
    float vv = 0.f;
#pragma unroll
    for (int j = 0; j < 4; ++j) { float d = y[j] - mu; vv += d * d; }
    red[t] = vv; __syncthreads();
    for (int o = 128; o > 0; o >>= 1) { if (t < o) red[t] += red[t + o]; __syncthreads(); }
    float var = red[0] / (float)DMODEL;
#pragma unroll
    for (int j = 0; j < 4; ++j) {
        float o = (y[j] - mu) / var;
        Xout[(size_t)row * DMODEL + t + 256 * j] = o;
        Xb[(size_t)row * DMODEL + t + 256 * j]  = (__bf16)o;
    }
}

// ---------------------------------------------------------------------------
// Last-token logits + row softmax
// ---------------------------------------------------------------------------
__global__ void logits_kernel(const float* __restrict__ X,
                              const float* __restrict__ Wout,
                              const float* __restrict__ bout,
                              float* __restrict__ logits) {
    int v = blockIdx.x * blockDim.x + threadIdx.x;
    int b = blockIdx.y;
    if (v >= VOCAB) return;
    const float* xr = X + ((size_t)(b * SEQ + SEQ - 1)) * DMODEL;
    float acc = bout[v];
    for (int d = 0; d < DMODEL; ++d) acc += xr[d] * Wout[(size_t)d * VOCAB + v];
    logits[(size_t)b * VOCAB + v] = acc;
}

__global__ __launch_bounds__(256) void softmax_kernel(
    const float* __restrict__ logits, float* __restrict__ out) {
    __shared__ float red[256];
    int b = blockIdx.x, t = threadIdx.x;
    const float* Lp = logits + (size_t)b * VOCAB;
    float m = -1e30f;
    for (int i = t; i < VOCAB; i += 256) m = fmaxf(m, Lp[i]);
    red[t] = m; __syncthreads();
    for (int o = 128; o > 0; o >>= 1) { if (t < o) red[t] = fmaxf(red[t], red[t + o]); __syncthreads(); }
    m = red[0]; __syncthreads();
    float s = 0.f;
    for (int i = t; i < VOCAB; i += 256) s += __expf(Lp[i] - m);
    red[t] = s; __syncthreads();
    for (int o = 128; o > 0; o >>= 1) { if (t < o) red[t] += red[t + o]; __syncthreads(); }
    s = red[0];
    for (int i = t; i < VOCAB; i += 256)
        out[(size_t)b * VOCAB + i] = __expf(Lp[i] - m) / s;
}

// ---------------------------------------------------------------------------
// Host orchestration
// ---------------------------------------------------------------------------
extern "C" void kernel_launch(void* const* d_in, const int* in_sizes, int n_in,
                              void* d_out, int out_size, void* d_ws, size_t ws_size,
                              hipStream_t stream) {
    const float* X    = (const float*)d_in[0];
    const float* Wq   = (const float*)d_in[1];
    const float* Wk   = (const float*)d_in[2];
    const float* Wv   = (const float*)d_in[3];
    const float* bq   = (const float*)d_in[4];
    const float* bk   = (const float*)d_in[5];
    const float* bv   = (const float*)d_in[6];
    const float* Wffn = (const float*)d_in[7];
    const float* bffn = (const float*)d_in[8];
    const float* Wout = (const float*)d_in[9];
    const float* bout = (const float*)d_in[10];
    float* out = (float*)d_out;

    const size_t NX = (size_t)MROWS * DMODEL;
    const size_t NQ = (size_t)HNUM * MROWS * DHEAD;

    char* ws = (char*)d_ws;
    size_t off = 0;
    auto carve = [&](size_t bytes) {
        void* p = ws + off;
        off = (off + bytes + 255) & ~(size_t)255;
        return p;
    };
    float*  Xcur = (float*)carve(NX * 4);
    __bf16* Xb   = (__bf16*)carve(NX * 2);
    __bf16* Qb   = (__bf16*)carve(NQ * 2);
    __bf16* Kb   = (__bf16*)carve(NQ * 2);
    __bf16* Vtb  = (__bf16*)carve(NQ * 2);
    float*  Zb   = (float*)carve(NX * 4);
    __bf16* WqT  = (__bf16*)carve((size_t)HNUM * DHEAD * DMODEL * 2);
    __bf16* WkT  = (__bf16*)carve((size_t)HNUM * DHEAD * DMODEL * 2);
    __bf16* WvT  = (__bf16*)carve((size_t)HNUM * DHEAD * DMODEL * 2);
    __bf16* WfT  = (__bf16*)carve((size_t)DMODEL * DMODEL * 2);
    float*  logb = (float*)carve((size_t)BATCH * VOCAB * 4);
    (void)ws_size; (void)in_sizes; (void)n_in; (void)out_size;

    x_init_kernel<<<(int)((NX + 255) / 256), 256, 0, stream>>>(X, Xcur, Xb, (int)NX);

    const float invSqrtDh = 0.125f;  // 1/sqrt(64), folded into Q
    const size_t wqkvLayer = (size_t)HNUM * DMODEL * DHEAD;

    for (int l = 0; l < LNUM; ++l) {
        transpose_cvt_kernel<<<dim3(128, HNUM), 256, 0, stream>>>(
            Wq + (size_t)l * wqkvLayer, WqT, DMODEL, DHEAD);
        transpose_cvt_kernel<<<dim3(128, HNUM), 256, 0, stream>>>(
            Wk + (size_t)l * wqkvLayer, WkT, DMODEL, DHEAD);
        transpose_cvt_kernel<<<dim3(128, HNUM), 256, 0, stream>>>(
            Wv + (size_t)l * wqkvLayer, WvT, DMODEL, DHEAD);
        transpose_cvt_kernel<<<dim3(256, 1), 256, 0, stream>>>(
            Wffn + (size_t)l * DMODEL * DMODEL, WfT, DMODEL, DMODEL);

        dim3 gq(MROWS / 128, DHEAD / 64, HNUM);
        gemm_bf16_kernel<MODE_QK><<<gq, 256, 0, stream>>>(
            Xb, WqT, bq + (size_t)l * HNUM * DHEAD, Qb,
            MROWS, DHEAD, DMODEL,
            (size_t)DHEAD * DMODEL, (size_t)MROWS * DHEAD, DHEAD, invSqrtDh);
        gemm_bf16_kernel<MODE_QK><<<gq, 256, 0, stream>>>(
            Xb, WkT, bk + (size_t)l * HNUM * DHEAD, Kb,
            MROWS, DHEAD, DMODEL,
            (size_t)DHEAD * DMODEL, (size_t)MROWS * DHEAD, DHEAD, 1.0f);
        gemm_bf16_kernel<MODE_V><<<gq, 256, 0, stream>>>(
            Xb, WvT, bv + (size_t)l * HNUM * DHEAD, Vtb,
            MROWS, DHEAD, DMODEL,
            (size_t)DHEAD * DMODEL, (size_t)DHEAD * MROWS, DHEAD, 1.0f);

        attn_kernel<<<HNUM * BATCH * (SEQ / 16), 32, 0, stream>>>(Qb, Kb, Vtb, Zb);
        add_norm_kernel<<<MROWS, 256, 0, stream>>>(Xcur, Zb, Xcur, Xb);

        dim3 gf(MROWS / 128, DMODEL / 64, 1);
        gemm_bf16_kernel<MODE_FFN><<<gf, 256, 0, stream>>>(
            Xb, WfT, bffn + (size_t)l * DMODEL, Zb,
            MROWS, DMODEL, DMODEL, 0, 0, 0, 1.0f);
        add_norm_kernel<<<MROWS, 256, 0, stream>>>(Xcur, Zb, Xcur, Xb);
    }

    logits_kernel<<<dim3((VOCAB + 255) / 256, BATCH), 256, 0, stream>>>(
        Xcur, Wout, bout, logb);
    softmax_kernel<<<BATCH, 256, 0, stream>>>(logb, out);
}
